// SelfAttention_77893526880753
// MI455X (gfx1250) — matrix operand order repaired and checked by reading, not compiled, not run on hardware
//
#include <hip/hip_runtime.h>

#define B_ 2
#define S_ 2048
#define D_ 1024
#define H_ 16
#define HD_ 64
#define BS_ (B_*S_)

#if defined(__AMDGCN__) && __has_builtin(__builtin_amdgcn_global_load_async_to_lds_b128) && __has_builtin(__builtin_amdgcn_s_wait_asynccnt)
#define HAVE_ASYNC 1
#else
#define HAVE_ASYNC 0
#endif

typedef __attribute__((ext_vector_type(16))) __bf16 v16bf;
typedef __attribute__((ext_vector_type(8)))  float  v8f;

union Frag { uint4 u[2]; v16bf v; };

__device__ __forceinline__ unsigned short f2bf(float f) {
  __bf16 h = (__bf16)f;                       // native v_cvt on gfx1250
  return __builtin_bit_cast(unsigned short, h);
}

__device__ __forceinline__ unsigned pk2bf(float a, float b) {
#if defined(__AMDGCN__) && __has_builtin(__builtin_amdgcn_cvt_pk_bf16_f32)
  typedef __attribute__((ext_vector_type(2))) __bf16 v2bf;
  v2bf t = __builtin_amdgcn_cvt_pk_bf16_f32(a, b);
  return __builtin_bit_cast(unsigned, t);
#else
  return (unsigned)f2bf(a) | ((unsigned)f2bf(b) << 16);
#endif
}

#if HAVE_ASYNC
typedef int v4i_ty __attribute__((vector_size(16)));
typedef __attribute__((address_space(1))) v4i_ty* gv4i_ptr;   // global v4i*
typedef __attribute__((address_space(3))) v4i_ty* lv4i_ptr;   // LDS v4i*
__device__ __forceinline__ void async_cp16(const void* g, void* l) {
  __builtin_amdgcn_global_load_async_to_lds_b128(
      (gv4i_ptr)(void*)g, (lv4i_ptr)l, 0, 0);
}
#endif

__device__ __forceinline__ void async_wait() {
#if HAVE_ASYNC
  __builtin_amdgcn_s_wait_asynccnt(0);
#endif
}

__device__ __forceinline__ v8f wmma_bf16(Frag a, Frag b, v8f c) {
  return __builtin_amdgcn_wmma_f32_16x16x32_bf16(false, a.v, false, b.v,
                                                 (short)0, c, false, false);
}

// ------------------------------------------------ transpose+convert weights
__global__ __launch_bounds__(256) void cvtT_kernel(const float* __restrict__ src,
                                                   unsigned short* __restrict__ dst) {
  __shared__ float t[32][33];
  const int k0 = blockIdx.x * 32, n0 = blockIdx.y * 32;
  const int tx = threadIdx.x & 31, ty = threadIdx.x >> 5;
  #pragma unroll
  for (int i = ty; i < 32; i += 8)
    t[i][tx] = src[(size_t)(k0 + i) * D_ + n0 + tx];
  __syncthreads();
  #pragma unroll
  for (int i = ty; i < 32; i += 8)
    dst[(size_t)(n0 + i) * D_ + k0 + tx] = f2bf(t[tx][i]);
}

// ------------------------------------------------ QKV GEMM + RoPE
// grid (BS_/128, 3*H_); 256 threads. 128x64 tile, K-step 32, double-buffered.
__global__ __launch_bounds__(256) void qkv_kernel(
    const float* __restrict__ x,
    const unsigned short* __restrict__ wqT,
    const unsigned short* __restrict__ wkT,
    const unsigned short* __restrict__ wvT,
    const float* __restrict__ fcos, const float* __restrict__ fsin,
    unsigned short* __restrict__ qw, unsigned short* __restrict__ kw,
    unsigned short* __restrict__ vwT)
{
  __shared__ alignas(16) unsigned short As[2][128 * 32];  // [m][k]
  __shared__ alignas(16) unsigned short Bs[2][64 * 32];   // [n][k]

  const int nblk = blockIdx.y;
  const int wsel = nblk >> 4;          // 0=q 1=k 2=v
  const int h    = nblk & 15;
  const unsigned short* WT = (wsel == 0) ? wqT : ((wsel == 1) ? wkT : wvT);

  const int m0 = blockIdx.x * 128;
  const int n0 = h * HD_;
  const int tid  = threadIdx.x;
  const int wave = tid >> 5;
  const int lane = tid & 31;
  const int half = lane >> 4;
  const int l16  = lane & 15;

  const v8f zero = {0.f,0.f,0.f,0.f,0.f,0.f,0.f,0.f};
  v8f acc[4] = {zero, zero, zero, zero};

  float4 areg[4];
  uint4  breg;
  const int ar = tid >> 3, ac4 = tid & 7;
  const int br = tid >> 2, bc4 = tid & 3;

  auto issue = [&](int kk, int buf) {
    #pragma unroll
    for (int it = 0; it < 4; ++it) {
      int r = ar + it * 32;
      areg[it] = *(const float4*)(x + (size_t)(m0 + r) * D_ + kk + ac4 * 4);
    }
#if HAVE_ASYNC
    async_cp16(WT + (size_t)(n0 + br) * D_ + kk + bc4 * 8,
               &Bs[buf][br * 32 + bc4 * 8]);
#else
    (void)buf;
    breg = *(const uint4*)(WT + (size_t)(n0 + br) * D_ + kk + bc4 * 8);
#endif
  };
  auto commit = [&](int buf) {
    #pragma unroll
    for (int it = 0; it < 4; ++it) {
      int r = ar + it * 32;
      uint2 p;
      p.x = pk2bf(areg[it].x, areg[it].y);
      p.y = pk2bf(areg[it].z, areg[it].w);
      *(uint2*)(&As[buf][r * 32 + ac4 * 4]) = p;
    }
#if !HAVE_ASYNC
    *(uint4*)(&Bs[buf][br * 32 + bc4 * 8]) = breg;
#endif
  };

  const int NS = D_ / 32;
  issue(0, 0);
  commit(0);
  async_wait();
  __syncthreads();

  for (int step = 0; step < NS; ++step) {
    const int cur = step & 1;
    if (step + 1 < NS) issue((step + 1) * 32, cur ^ 1);

    Frag fa;
    fa.u[0] = *(const uint4*)(&As[cur][(wave * 16 + l16) * 32 + 16 * half]);
    fa.u[1] = *(const uint4*)(&As[cur][(wave * 16 + l16) * 32 + 16 * half + 8]);
    Frag fb0, fb1, fb2, fb3;
    fb0.u[0] = *(const uint4*)(&Bs[cur][( 0 + l16) * 32 + 16 * half]);
    fb0.u[1] = *(const uint4*)(&Bs[cur][( 0 + l16) * 32 + 16 * half + 8]);
    fb1.u[0] = *(const uint4*)(&Bs[cur][(16 + l16) * 32 + 16 * half]);
    fb1.u[1] = *(const uint4*)(&Bs[cur][(16 + l16) * 32 + 16 * half + 8]);
    fb2.u[0] = *(const uint4*)(&Bs[cur][(32 + l16) * 32 + 16 * half]);
    fb2.u[1] = *(const uint4*)(&Bs[cur][(32 + l16) * 32 + 16 * half + 8]);
    fb3.u[0] = *(const uint4*)(&Bs[cur][(48 + l16) * 32 + 16 * half]);
    fb3.u[1] = *(const uint4*)(&Bs[cur][(48 + l16) * 32 + 16 * half + 8]);
    acc[0] = wmma_bf16(fa, fb0, acc[0]);
    acc[1] = wmma_bf16(fa, fb1, acc[1]);
    acc[2] = wmma_bf16(fa, fb2, acc[2]);
    acc[3] = wmma_bf16(fa, fb3, acc[3]);

    if (step + 1 < NS) commit(cur ^ 1);
    async_wait();
    __syncthreads();
  }

  const int mrow0 = m0 + wave * 16 + 8 * half;
  const int b = m0 >> 11;
  const int srow_blk = m0 & (S_ - 1);
  if (wsel < 2) {
    // RoPE (q pre-scaled by 1/sqrt(HD)); stage tile in LDS, store coalesced.
    unsigned short* outw = (wsel == 0) ? qw : kw;
    unsigned short* stg = &As[0][0];                 // 128*64 bf16 = 16KB
    const float qscale = (wsel == 0) ? 0.125f : 1.0f;
    #pragma unroll
    for (int t = 0; t < 4; ++t) {
      const int hd  = t * 16 + l16;
      const int p   = hd >> 1;
      const int odd = hd & 1;
      #pragma unroll
      for (int i = 0; i < 8; ++i) {
        float v  = acc[t][i];
        float pv = __shfl_xor(v, 1, 32);
        int s = (mrow0 & (S_ - 1)) + i;
        float c  = fcos[s * (HD_/2) + p];
        float sn = fsin[s * (HD_/2) + p];
        float o = (odd ? (pv * sn + v * c) : (v * c - pv * sn)) * qscale;
        stg[(wave * 16 + 8 * half + i) * 64 + hd] = f2bf(o);
      }
    }
    __syncthreads();
    const size_t base = ((size_t)(b * H_ + h) * S_ + srow_blk) * HD_;
    #pragma unroll
    for (int it = 0; it < 4; ++it) {
      int idx = tid + it * 256;                      // 1024 x uint4
      *(uint4*)(outw + base + idx * 8) = *(const uint4*)(stg + idx * 8);
    }
  } else {
    // V stored transposed [b*H+h][hd][s]; 8 consecutive s -> one uint4
    const int srow0 = mrow0 & (S_ - 1);
    #pragma unroll
    for (int t = 0; t < 4; ++t) {
      const int hd = t * 16 + l16;
      uint4 pv;
      pv.x = pk2bf(acc[t][0], acc[t][1]);
      pv.y = pk2bf(acc[t][2], acc[t][3]);
      pv.z = pk2bf(acc[t][4], acc[t][5]);
      pv.w = pk2bf(acc[t][6], acc[t][7]);
      *(uint4*)(vwT + ((size_t)(b * H_ + h) * HD_ + hd) * S_ + srow0) = pv;
    }
  }
}

// ------------------------------------------------ attention (flash-style)
// grid (S_/128, B_*H_); 256 threads. Double-buffered K/V chunks with async
// prefetch; Q fragments loaded once from global; online softmax.
__global__ __launch_bounds__(256) void attn_kernel(
    const unsigned short* __restrict__ qw,
    const unsigned short* __restrict__ kw,
    const unsigned short* __restrict__ vwT,
    unsigned short* __restrict__ attnw)
{
  __shared__ alignas(16) unsigned short Ks [2][64 * 64];  // [kv][hd]
  __shared__ alignas(16) unsigned short Vts[2][64 * 64];  // [hd][kv]
  __shared__ alignas(16) unsigned short Ps [128 * 64];    // [qrow][kv]

  const int bh = blockIdx.y;
  const unsigned short* qp = qw  + (size_t)bh * S_ * HD_;
  const unsigned short* kp = kw  + (size_t)bh * S_ * HD_;
  const unsigned short* vp = vwT + (size_t)bh * HD_ * S_;

  const int s0   = blockIdx.x * 128;
  const int tid  = threadIdx.x;
  const int wave = tid >> 5;
  const int lane = tid & 31;
  const int half = lane >> 4;
  const int l16  = lane & 15;
  const int r0   = wave * 16;

  Frag qf[2];
  #pragma unroll
  for (int ks2 = 0; ks2 < 2; ++ks2) {
    qf[ks2].u[0] = *(const uint4*)(qp + (size_t)(s0 + r0 + l16) * HD_ + ks2 * 32 + 16 * half);
    qf[ks2].u[1] = *(const uint4*)(qp + (size_t)(s0 + r0 + l16) * HD_ + ks2 * 32 + 16 * half + 8);
  }

  const v8f zero = {0.f,0.f,0.f,0.f,0.f,0.f,0.f,0.f};
  v8f o[4] = {zero, zero, zero, zero};
  float mi[8], li[8];
  #pragma unroll
  for (int i = 0; i < 8; ++i) { mi[i] = -1e30f; li[i] = 0.f; }

  const int cr = tid >> 3, cc8 = tid & 7;
  uint4 kreg[2], vreg[2];
  auto issueKV = [&](int kc, int buf) {
    #pragma unroll
    for (int it = 0; it < 2; ++it) {
      int r = cr + it * 32;
#if HAVE_ASYNC
      async_cp16(kp + (size_t)(kc + r) * HD_ + cc8 * 8, &Ks[buf][r * 64 + cc8 * 8]);
      async_cp16(vp + (size_t)r * S_ + kc + cc8 * 8,   &Vts[buf][r * 64 + cc8 * 8]);
#else
      (void)buf;
      kreg[it] = *(const uint4*)(kp + (size_t)(kc + r) * HD_ + cc8 * 8);
      vreg[it] = *(const uint4*)(vp + (size_t)r * S_ + kc + cc8 * 8);
#endif
    }
  };
  auto commitKV = [&](int buf) {
#if !HAVE_ASYNC
    #pragma unroll
    for (int it = 0; it < 2; ++it) {
      int r = cr + it * 32;
      *(uint4*)(&Ks[buf][r * 64 + cc8 * 8])  = kreg[it];
      *(uint4*)(&Vts[buf][r * 64 + cc8 * 8]) = vreg[it];
    }
#else
    (void)buf;
#endif
  };

  const int NC = S_ / 64;
  issueKV(0, 0);
  commitKV(0);
  async_wait();
  __syncthreads();

  for (int it = 0; it < NC; ++it) {
    const int cur = it & 1;
    if (it + 1 < NC) issueKV((it + 1) * 64, cur ^ 1);

    // scores = Q @ K^T (q already scaled)
    v8f sc[4] = {zero, zero, zero, zero};
    #pragma unroll
    for (int ks2 = 0; ks2 < 2; ++ks2) {
      const int ks = ks2 * 32;
      Frag fb0, fb1, fb2, fb3;
      fb0.u[0] = *(const uint4*)(&Ks[cur][( 0 + l16) * 64 + ks + 16 * half]);
      fb0.u[1] = *(const uint4*)(&Ks[cur][( 0 + l16) * 64 + ks + 16 * half + 8]);
      fb1.u[0] = *(const uint4*)(&Ks[cur][(16 + l16) * 64 + ks + 16 * half]);
      fb1.u[1] = *(const uint4*)(&Ks[cur][(16 + l16) * 64 + ks + 16 * half + 8]);
      fb2.u[0] = *(const uint4*)(&Ks[cur][(32 + l16) * 64 + ks + 16 * half]);
      fb2.u[1] = *(const uint4*)(&Ks[cur][(32 + l16) * 64 + ks + 16 * half + 8]);
      fb3.u[0] = *(const uint4*)(&Ks[cur][(48 + l16) * 64 + ks + 16 * half]);
      fb3.u[1] = *(const uint4*)(&Ks[cur][(48 + l16) * 64 + ks + 16 * half + 8]);
      sc[0] = wmma_bf16(qf[ks2], fb0, sc[0]);
      sc[1] = wmma_bf16(qf[ks2], fb1, sc[1]);
      sc[2] = wmma_bf16(qf[ks2], fb2, sc[2]);
      sc[3] = wmma_bf16(qf[ks2], fb3, sc[3]);
    }

    // online softmax
    #pragma unroll
    for (int i = 0; i < 8; ++i) {
      float v0 = sc[0][i], v1 = sc[1][i], v2 = sc[2][i], v3 = sc[3][i];
      float rmax = fmaxf(fmaxf(v0, v1), fmaxf(v2, v3));
      rmax = fmaxf(rmax, __shfl_xor(rmax, 1, 32));
      rmax = fmaxf(rmax, __shfl_xor(rmax, 2, 32));
      rmax = fmaxf(rmax, __shfl_xor(rmax, 4, 32));
      rmax = fmaxf(rmax, __shfl_xor(rmax, 8, 32));
      float mnew  = fmaxf(mi[i], rmax);
      float alpha = __expf(mi[i] - mnew);
      float p0 = __expf(v0 - mnew), p1 = __expf(v1 - mnew);
      float p2 = __expf(v2 - mnew), p3 = __expf(v3 - mnew);
      int row = r0 + 8 * half + i;
      Ps[row * 64 +  0 + l16] = f2bf(p0);
      Ps[row * 64 + 16 + l16] = f2bf(p1);
      Ps[row * 64 + 32 + l16] = f2bf(p2);
      Ps[row * 64 + 48 + l16] = f2bf(p3);
      float rsum = p0 + p1 + p2 + p3;
      rsum += __shfl_xor(rsum, 1, 32);
      rsum += __shfl_xor(rsum, 2, 32);
      rsum += __shfl_xor(rsum, 4, 32);
      rsum += __shfl_xor(rsum, 8, 32);
      li[i] = li[i] * alpha + rsum;
      mi[i] = mnew;
      o[0][i] *= alpha; o[1][i] *= alpha; o[2][i] *= alpha; o[3][i] *= alpha;
    }
    __syncthreads();

    // O += P @ V
    #pragma unroll
    for (int ks2 = 0; ks2 < 2; ++ks2) {
      const int ks = ks2 * 32;
      Frag fa;
      fa.u[0] = *(const uint4*)(Ps + (r0 + l16) * 64 + ks + 16 * half);
      fa.u[1] = *(const uint4*)(Ps + (r0 + l16) * 64 + ks + 16 * half + 8);
      Frag fb0, fb1, fb2, fb3;
      fb0.u[0] = *(const uint4*)(&Vts[cur][( 0 + l16) * 64 + ks + 16 * half]);
      fb0.u[1] = *(const uint4*)(&Vts[cur][( 0 + l16) * 64 + ks + 16 * half + 8]);
      fb1.u[0] = *(const uint4*)(&Vts[cur][(16 + l16) * 64 + ks + 16 * half]);
      fb1.u[1] = *(const uint4*)(&Vts[cur][(16 + l16) * 64 + ks + 16 * half + 8]);
      fb2.u[0] = *(const uint4*)(&Vts[cur][(32 + l16) * 64 + ks + 16 * half]);
      fb2.u[1] = *(const uint4*)(&Vts[cur][(32 + l16) * 64 + ks + 16 * half + 8]);
      fb3.u[0] = *(const uint4*)(&Vts[cur][(48 + l16) * 64 + ks + 16 * half]);
      fb3.u[1] = *(const uint4*)(&Vts[cur][(48 + l16) * 64 + ks + 16 * half + 8]);
      o[0] = wmma_bf16(fa, fb0, o[0]);
      o[1] = wmma_bf16(fa, fb1, o[1]);
      o[2] = wmma_bf16(fa, fb2, o[2]);
      o[3] = wmma_bf16(fa, fb3, o[3]);
    }

    if (it + 1 < NC) commitKV(cur ^ 1);
    async_wait();
    __syncthreads();
  }

  // epilogue: stage in LDS (Ps is free), then coalesced b128 stores
  const int b = bh >> 4, h = bh & 15;
  unsigned short* stg = Ps;
  #pragma unroll
  for (int i = 0; i < 8; ++i) {
    float inv = 1.0f / li[i];
    int row = r0 + 8 * half + i;
    stg[row * 64 +  0 + l16] = f2bf(o[0][i] * inv);
    stg[row * 64 + 16 + l16] = f2bf(o[1][i] * inv);
    stg[row * 64 + 32 + l16] = f2bf(o[2][i] * inv);
    stg[row * 64 + 48 + l16] = f2bf(o[3][i] * inv);
  }
  __syncthreads();
  #pragma unroll
  for (int it = 0; it < 4; ++it) {
    int idx = tid + it * 256;
    int r = idx >> 3, c8 = idx & 7;
    *(uint4*)(attnw + (size_t)(b * S_ + s0 + r) * D_ + h * HD_ + c8 * 8) =
        *(const uint4*)(stg + idx * 8);
  }
}

// ------------------------------------------------ output projection
__global__ __launch_bounds__(256) void oproj_kernel(
    const unsigned short* __restrict__ attnw,
    const unsigned short* __restrict__ woT,
    float* __restrict__ out)
{
  __shared__ alignas(16) unsigned short As[2][128 * 32];
  __shared__ alignas(16) unsigned short Bs[2][64 * 32];

  const int m0 = blockIdx.x * 128;
  const int n0 = blockIdx.y * 64;
  const int tid  = threadIdx.x;
  const int wave = tid >> 5;
  const int lane = tid & 31;
  const int half = lane >> 4;
  const int l16  = lane & 15;

  const v8f zero = {0.f,0.f,0.f,0.f,0.f,0.f,0.f,0.f};
  v8f acc[4] = {zero, zero, zero, zero};

  uint4 areg[2], breg;
  const int ar = tid >> 2, ac4 = tid & 3;
  auto issue = [&](int kk, int buf) {
#if HAVE_ASYNC
    #pragma unroll
    for (int it = 0; it < 2; ++it) {
      int r = ar + it * 64;
      async_cp16(attnw + (size_t)(m0 + r) * D_ + kk + ac4 * 8,
                 &As[buf][r * 32 + ac4 * 8]);
    }
    async_cp16(woT + (size_t)(n0 + ar) * D_ + kk + ac4 * 8,
               &Bs[buf][ar * 32 + ac4 * 8]);
#else
    (void)buf;
    #pragma unroll
    for (int it = 0; it < 2; ++it) {
      int r = ar + it * 64;
      areg[it] = *(const uint4*)(attnw + (size_t)(m0 + r) * D_ + kk + ac4 * 8);
    }
    breg = *(const uint4*)(woT + (size_t)(n0 + ar) * D_ + kk + ac4 * 8);
#endif
  };
  auto commit = [&](int buf) {
#if !HAVE_ASYNC
    #pragma unroll
    for (int it = 0; it < 2; ++it) {
      int r = ar + it * 64;
      *(uint4*)(&As[buf][r * 32 + ac4 * 8]) = areg[it];
    }
    *(uint4*)(&Bs[buf][ar * 32 + ac4 * 8]) = breg;
#else
    (void)buf;
#endif
  };

  const int NS = D_ / 32;
  issue(0, 0);
  commit(0);
  async_wait();
  __syncthreads();

  for (int step = 0; step < NS; ++step) {
    const int cur = step & 1;
    if (step + 1 < NS) issue((step + 1) * 32, cur ^ 1);

    Frag fa;
    fa.u[0] = *(const uint4*)(&As[cur][(wave * 16 + l16) * 32 + 16 * half]);
    fa.u[1] = *(const uint4*)(&As[cur][(wave * 16 + l16) * 32 + 16 * half + 8]);
    Frag fb0, fb1, fb2, fb3;
    fb0.u[0] = *(const uint4*)(&Bs[cur][( 0 + l16) * 32 + 16 * half]);
    fb0.u[1] = *(const uint4*)(&Bs[cur][( 0 + l16) * 32 + 16 * half + 8]);
    fb1.u[0] = *(const uint4*)(&Bs[cur][(16 + l16) * 32 + 16 * half]);
    fb1.u[1] = *(const uint4*)(&Bs[cur][(16 + l16) * 32 + 16 * half + 8]);
    fb2.u[0] = *(const uint4*)(&Bs[cur][(32 + l16) * 32 + 16 * half]);
    fb2.u[1] = *(const uint4*)(&Bs[cur][(32 + l16) * 32 + 16 * half + 8]);
    fb3.u[0] = *(const uint4*)(&Bs[cur][(48 + l16) * 32 + 16 * half]);
    fb3.u[1] = *(const uint4*)(&Bs[cur][(48 + l16) * 32 + 16 * half + 8]);
    acc[0] = wmma_bf16(fa, fb0, acc[0]);
    acc[1] = wmma_bf16(fa, fb1, acc[1]);
    acc[2] = wmma_bf16(fa, fb2, acc[2]);
    acc[3] = wmma_bf16(fa, fb3, acc[3]);

    if (step + 1 < NS) commit(cur ^ 1);
    async_wait();
    __syncthreads();
  }

  const int mrow0 = m0 + wave * 16 + 8 * half;
  #pragma unroll
  for (int t = 0; t < 4; ++t) {
    #pragma unroll
    for (int i = 0; i < 8; ++i) {
      out[(size_t)(mrow0 + i) * D_ + n0 + t * 16 + l16] = acc[t][i];
    }
  }
}

// ------------------------------------------------ launch
extern "C" void kernel_launch(void* const* d_in, const int* in_sizes, int n_in,
                              void* d_out, int out_size, void* d_ws, size_t ws_size,
                              hipStream_t stream) {
  const float* x  = (const float*)d_in[0];
  const float* fc = (const float*)d_in[1];
  const float* fs = (const float*)d_in[2];
  const float* wq = (const float*)d_in[3];
  const float* wk = (const float*)d_in[4];
  const float* wv = (const float*)d_in[5];
  const float* wo = (const float*)d_in[6];
  float* out = (float*)d_out;

  unsigned short* wqT  = (unsigned short*)d_ws;
  unsigned short* wkT  = wqT + (size_t)D_ * D_;
  unsigned short* wvT  = wkT + (size_t)D_ * D_;
  unsigned short* woT  = wvT + (size_t)D_ * D_;
  unsigned short* qwv  = woT + (size_t)D_ * D_;            // [B*H][S][HD]
  unsigned short* kwv  = qwv + (size_t)B_ * H_ * S_ * HD_;
  unsigned short* vwT  = kwv + (size_t)B_ * H_ * S_ * HD_; // [B*H][HD][S]
  unsigned short* attn = vwT + (size_t)B_ * H_ * S_ * HD_; // [BS_][D_]

  dim3 tgrid(D_ / 32, D_ / 32);
  cvtT_kernel<<<tgrid, 256, 0, stream>>>(wq, wqT);
  cvtT_kernel<<<tgrid, 256, 0, stream>>>(wk, wkT);
  cvtT_kernel<<<tgrid, 256, 0, stream>>>(wv, wvT);
  cvtT_kernel<<<tgrid, 256, 0, stream>>>(wo, woT);

  qkv_kernel<<<dim3(BS_ / 128, 3 * H_), 256, 0, stream>>>(
      x, wqT, wkT, wvT, fc, fs, qwv, kwv, vwT);

  attn_kernel<<<dim3(S_ / 128, B_ * H_), 256, 0, stream>>>(qwv, kwv, vwT, attn);

  oproj_kernel<<<dim3(BS_ / 128, D_ / 64), 256, 0, stream>>>(attn, woT, out);
}
